// PullAttention_64639257804905
// MI455X (gfx1250) — matrix operand order repaired and checked by reading, not compiled, not run on hardware
//
#include <hip/hip_runtime.h>
#include <hip/hip_bf16.h>
#include <math.h>

typedef __attribute__((ext_vector_type(16))) _Float16 v16h;
typedef __attribute__((ext_vector_type(8)))  _Float16 v8h;
typedef __attribute__((ext_vector_type(8)))  float    v8f;

#define B_    4
#define L_    2048
#define H_    8
#define E_    64
#define BM_   128
#define BN_   64
#define ALPHA_ 0.1f
#define SCALE_ 0.125f     // 1/sqrt(64)
#define FNEG_  -3.0e38f

// ---- WMMA fragment gathers (layouts per CDNA5 ISA 7.12.2, wave32) ----
// A-matrix 16x32 f16: lanes 0-15 row M=lane, halfs = K {0..7}+{16..23};
//                     lanes16-31 row M=lane-16, halfs = K {8..15}+{24..31}.
__device__ __forceinline__ v16h fragA(const _Float16* base, int row, int kb, int hi) {
  const _Float16* p = base + row * 64 + kb + hi * 8;
  v8h lo = *(const v8h*)p;
  v8h hh = *(const v8h*)(p + 16);
  return __builtin_shufflevector(lo, hh, 0,1,2,3,4,5,6,7,8,9,10,11,12,13,14,15);
}
// B-matrix 32x16 f16 (rows K, cols N), gathered from a transposed tile
// stored [n][k] row-major (64 cols): lanes 0-15 col N=lane, K = kb+{0..7}+{16..23};
//                                    lanes16-31 col N=lane-16, K = kb+{8..15}+{24..31}.
__device__ __forceinline__ v16h fragB(const _Float16* base, int nrow, int kb, int hi) {
  const _Float16* p = base + nrow * 64 + kb + hi * 8;
  v8h lo = *(const v8h*)p;
  v8h hh = *(const v8h*)(p + 16);
  return __builtin_shufflevector(lo, hh, 0,1,2,3,4,5,6,7,8,9,10,11,12,13,14,15);
}

__global__ __launch_bounds__(256)
void PullAttention_64639257804905_kernel(const float* __restrict__ Q,
                                         const float* __restrict__ K,
                                         const float* __restrict__ V,
                                         float* __restrict__ Out) {
  __shared__ _Float16 Kt[64 * 64];        // K tile, row-major [s][e]  (8 KB)
  __shared__ _Float16 VtT[64 * 64];       // V tile, transposed [d][s] (8 KB)
  __shared__ _Float16 Pst[8 * 16 * 64];   // per-wave Q/prob staging  (16 KB)

  const int tid  = threadIdx.x;
  const int lane = tid & 31;
  const int wid  = tid >> 5;
  const int nl   = lane & 15;
  const int hi   = lane >> 4;
  const int m0   = blockIdx.x * BM_;
  const int h    = blockIdx.y;
  const int b    = blockIdx.z;
  const int wmin = m0 + wid * 16;

  const v8f vzero = {0.f,0.f,0.f,0.f,0.f,0.f,0.f,0.f};

  // ---------- stage projected Q (f16) into this wave's LDS area ----------
  _Float16* qstage = Pst + wid * (16 * 64);
  {
    const int qrow  = lane >> 1;           // 0..15
    const int qhalf = lane & 1;            // elems [32*qhalf, +32)
    const size_t qoff = (((size_t)b * L_ + (wmin + qrow)) * H_ + h) * (size_t)E_ + qhalf * 32;
    const float4* q4 = (const float4*)(Q + qoff);
    float x[32];
    float amax = 0.f;
    #pragma unroll
    for (int i = 0; i < 8; ++i) {
      float4 t = q4[i];
      x[4*i+0] = t.x; x[4*i+1] = t.y; x[4*i+2] = t.z; x[4*i+3] = t.w;
    }
    #pragma unroll
    for (int i = 0; i < 32; ++i) amax = fmaxf(amax, fabsf(x[i]));
    amax = fmaxf(amax, __shfl_xor(amax, 1));
    const float thr = ALPHA_ * amax;
    #pragma unroll
    for (int i = 0; i < 32; ++i) {
      float v = (fabsf(x[i]) >= thr) ? x[i] : 0.f;
      qstage[qrow * 64 + qhalf * 32 + i] = (_Float16)v;
    }
  }
  v16h aQ0 = fragA(qstage, nl, 0,  hi);   // K-dim e = 0..31
  v16h aQ1 = fragA(qstage, nl, 32, hi);   // K-dim e = 32..63

  // ---------- flash accumulators ----------
  v8f o[4]; 
  #pragma unroll
  for (int j = 0; j < 4; ++j) o[j] = vzero;
  float mrun[8], ssum[8];
  #pragma unroll
  for (int r = 0; r < 8; ++r) { mrun[r] = FNEG_; ssum[r] = 0.f; }

  const int n_tiles = m0 / BN_ + 2;                  // causal frontier
  const int krow = tid >> 2, kseg = tid & 3;         // 64 rows x 4 quarter-rows
  const size_t kvbase = (((size_t)b * L_ + krow) * H_ + h) * (size_t)E_ + kseg * 16;
  const size_t srowstride = (size_t)H_ * E_;

  for (int t = 0; t < n_tiles; ++t) {
    const int s0 = t * BN_;

    // ---------- cooperative tile load: project K, transpose-convert V ----------
    {
      const float4* k4 = (const float4*)(K + kvbase + (size_t)s0 * srowstride);
      float x[16];
      float amax = 0.f;
      #pragma unroll
      for (int i = 0; i < 4; ++i) {
        float4 tt = k4[i];
        x[4*i+0] = tt.x; x[4*i+1] = tt.y; x[4*i+2] = tt.z; x[4*i+3] = tt.w;
      }
      #pragma unroll
      for (int i = 0; i < 16; ++i) amax = fmaxf(amax, fabsf(x[i]));
      amax = fmaxf(amax, __shfl_xor(amax, 1));
      amax = fmaxf(amax, __shfl_xor(amax, 2));
      const float thr = ALPHA_ * amax;
      #pragma unroll
      for (int i = 0; i < 16; ++i) {
        float v = (fabsf(x[i]) >= thr) ? x[i] : 0.f;
        Kt[krow * 64 + kseg * 16 + i] = (_Float16)v;
      }
      const float4* v4 = (const float4*)(V + kvbase + (size_t)s0 * srowstride);
      #pragma unroll
      for (int i = 0; i < 4; ++i) {
        float4 tt = v4[i];
        VtT[(kseg * 16 + 4*i + 0) * 64 + krow] = (_Float16)tt.x;
        VtT[(kseg * 16 + 4*i + 1) * 64 + krow] = (_Float16)tt.y;
        VtT[(kseg * 16 + 4*i + 2) * 64 + krow] = (_Float16)tt.z;
        VtT[(kseg * 16 + 4*i + 3) * 64 + krow] = (_Float16)tt.w;
      }
      if (t + 1 < n_tiles) {   // global_prefetch_b8 for next tile
        __builtin_prefetch(K + kvbase + (size_t)(s0 + BN_) * srowstride, 0, 1);
        __builtin_prefetch(V + kvbase + (size_t)(s0 + BN_) * srowstride, 0, 1);
      }
    }
    __syncthreads();

    // ---------- per-wave compute (skip fully-masked tiles) ----------
    if (s0 <= wmin + 15) {
      // scores: Q (16x64) * K^T (64x64) -> 4 tiles of 16x16 f32
      v8f cs[4];
      #pragma unroll
      for (int j = 0; j < 4; ++j) {
        v16h b0 = fragB(Kt, j * 16 + nl, 0,  hi);
        v16h b1 = fragB(Kt, j * 16 + nl, 32, hi);
        v8f c = vzero;
        c = __builtin_amdgcn_wmma_f32_16x16x32_f16(false, aQ0, false, b0, (short)0, c, false, false);
        c = __builtin_amdgcn_wmma_f32_16x16x32_f16(false, aQ1, false, b1, (short)0, c, false, false);
        cs[j] = c;
      }

      // scale + causal mask + online softmax
      const bool need_mask = (s0 + BN_ - 1 > wmin);
      float rmax[8];
      #pragma unroll
      for (int r = 0; r < 8; ++r) rmax[r] = FNEG_;
      #pragma unroll
      for (int j = 0; j < 4; ++j) {
        #pragma unroll
        for (int r = 0; r < 8; ++r) {
          float x = cs[j][r] * SCALE_;
          if (need_mask) {
            const int sg = s0 + j * 16 + nl;
            const int mg = wmin + r + 8 * hi;
            if (sg > mg) x = FNEG_;
          }
          cs[j][r] = x;
          rmax[r] = fmaxf(rmax[r], x);
        }
      }
      #pragma unroll
      for (int r = 0; r < 8; ++r) {
        float m = rmax[r];
        m = fmaxf(m, __shfl_xor(m, 1));
        m = fmaxf(m, __shfl_xor(m, 2));
        m = fmaxf(m, __shfl_xor(m, 4));
        m = fmaxf(m, __shfl_xor(m, 8));
        const float mnew  = fmaxf(mrun[r], m);
        const float alpha = __expf(mrun[r] - mnew);
        mrun[r] = mnew;
        ssum[r] *= alpha;
        o[0][r] *= alpha; o[1][r] *= alpha; o[2][r] *= alpha; o[3][r] *= alpha;
      }

      // exponentiate, row-sum, stage probs (C-layout scatter -> A-layout gather)
      _Float16* pst = Pst + wid * (16 * 64);
      float lsum[8];
      #pragma unroll
      for (int r = 0; r < 8; ++r) lsum[r] = 0.f;
      #pragma unroll
      for (int j = 0; j < 4; ++j) {
        #pragma unroll
        for (int r = 0; r < 8; ++r) {
          const float p = __expf(cs[j][r] - mrun[r]);
          lsum[r] += p;
          pst[(r + 8 * hi) * 64 + j * 16 + nl] = (_Float16)p;
        }
      }
      #pragma unroll
      for (int r = 0; r < 8; ++r) {
        float s = lsum[r];
        s += __shfl_xor(s, 1);
        s += __shfl_xor(s, 2);
        s += __shfl_xor(s, 4);
        s += __shfl_xor(s, 8);
        ssum[r] += s;
      }

      // O += P (16x64) * V (64x64)
      v16h aP0 = fragA(pst, nl, 0,  hi);
      v16h aP1 = fragA(pst, nl, 32, hi);
      #pragma unroll
      for (int jd = 0; jd < 4; ++jd) {
        v16h b0 = fragB(VtT, jd * 16 + nl, 0,  hi);
        v16h b1 = fragB(VtT, jd * 16 + nl, 32, hi);
        o[jd] = __builtin_amdgcn_wmma_f32_16x16x32_f16(false, aP0, false, b0, (short)0, o[jd], false, false);
        o[jd] = __builtin_amdgcn_wmma_f32_16x16x32_f16(false, aP1, false, b1, (short)0, o[jd], false, false);
      }
    }
    __syncthreads();
  }

  // ---------- epilogue: normalize, write [B,L,H,D] f32 ----------
  #pragma unroll
  for (int r = 0; r < 8; ++r) {
    const float inv = 1.0f / ssum[r];
    const int mg = wmin + r + 8 * hi;
    float* orow = Out + (((size_t)b * L_ + mg) * H_ + h) * (size_t)E_;
    #pragma unroll
    for (int jd = 0; jd < 4; ++jd) {
      orow[jd * 16 + nl] = o[jd][r] * inv;
    }
  }
}

extern "C" void kernel_launch(void* const* d_in, const int* in_sizes, int n_in,
                              void* d_out, int out_size, void* d_ws, size_t ws_size,
                              hipStream_t stream) {
  (void)in_sizes; (void)n_in; (void)out_size; (void)d_ws; (void)ws_size;
  const float* Q = (const float*)d_in[0];   // [B,L,H,E] f32
  const float* K = (const float*)d_in[1];   // [B,S,H,E] f32
  const float* V = (const float*)d_in[2];   // [B,S,H,D] f32
  // d_in[3] = causal mask (bool) — causality is computed analytically.
  float* Out = (float*)d_out;               // [B,L,H,D] f32

  dim3 grid(L_ / BM_, H_, B_);
  dim3 block(256);
  PullAttention_64639257804905_kernel<<<grid, block, 0, stream>>>(Q, K, V, Out);
}
